// SOC_45122926411976
// MI455X (gfx1250) — matrix operand order repaired and checked by reading, not compiled, not run on hardware
//
#include <hip/hip_runtime.h>

typedef _Float16 v16h __attribute__((ext_vector_type(16)));
typedef _Float16 v8h  __attribute__((ext_vector_type(8)));
typedef float    v8f  __attribute__((ext_vector_type(8)));

#define CCH 32
#define KW  3
#define KG  11   // effective kernel after conv-exp (K=3, 5 iters)

// ---------------- workspace layout (floats / bytes) ----------------
#define OFF_KN   0                       // 9216 floats: normalized skew kernel
#define OFF_KIA  9216                    // 123904 floats (ping)
#define OFF_KIB  133120                  // 123904 floats (pong)
#define OFF_KG   257024                  // 123904 floats (32x32x11x11)
#define PW_BYTE_OFF 1523712u             // 123904 halves packed A-fragments
#define XT_BYTE_OFF 1771520u             // 16*256*256*32 halves NHWC-f16

// ======================= stage 1: skew ============================
__global__ void skew_k(const float* __restrict__ w, float* __restrict__ ks) {
    int idx = blockIdx.x * 256 + threadIdx.x;
    if (idx >= CCH*CCH*KW*KW) return;
    int q = idx % 3, p = (idx/3) % 3, c = (idx/9) % 32, o = idx / 288;
    ks[idx] = w[idx] - w[((c*32 + o)*3 + (2-p))*3 + (2-q)];
}

// ================= stage 2: fantastic four (1 block) ==============
__device__ float block_sum(float v, float* red) {
    int t = threadIdx.x;
    red[t] = v; __syncthreads();
    for (int s = 128; s > 0; s >>= 1) { if (t < s) red[t] += red[t+s]; __syncthreads(); }
    float r = red[0]; __syncthreads(); return r;
}
__device__ void l2norm(float* a, int n, float* red) {
    int t = threadIdx.x; float s = 0.f;
    for (int i = t; i < n; i += 256) s += a[i]*a[i];
    float tot = block_sum(s, red);
    float inv = 1.f / (sqrtf(tot) + 1e-12f);
    for (int i = t; i < n; i += 256) a[i] *= inv;
    __syncthreads();
}

__global__ __launch_bounds__(256) void ff_kernel(const float* __restrict__ ksin,
        const float* __restrict__ gu1, const float* __restrict__ gu2,
        const float* __restrict__ gu3, const float* __restrict__ gu4,
        float* __restrict__ knout) {
    __shared__ float k[9216];
    __shared__ float u1[96], v1[96], u2[96], v2[96];
    __shared__ float u3[288], v3[32], u4[288], v4[32];
    __shared__ float red[256];
    int t = threadIdx.x;
    for (int i = t; i < 9216; i += 256) k[i] = ksin[i];
    for (int i = t; i < 96;  i += 256) { u1[i] = gu1[i]; u2[i] = gu2[i]; }
    for (int i = t; i < 288; i += 256) { u3[i] = gu3[i]; u4[i] = gu4[i]; }
    __syncthreads();
    l2norm(u1, 96, red); l2norm(u2, 96, red); l2norm(u3, 288, red); l2norm(u4, 288, red);

    for (int it = 0; it < 3; ++it) {
        for (int i = t; i < 96; i += 256) { int o = i/3, p = i%3; float s = 0.f;
            for (int c = 0; c < 32; ++c) for (int q = 0; q < 3; ++q)
                s += k[((o*32+c)*3+p)*3+q] * u1[c*3+q];
            v1[i] = s; }
        __syncthreads(); l2norm(v1, 96, red);
        for (int i = t; i < 96; i += 256) { int c = i/3, q = i%3; float s = 0.f;
            for (int o = 0; o < 32; ++o) for (int p = 0; p < 3; ++p)
                s += k[((o*32+c)*3+p)*3+q] * v1[o*3+p];
            u1[i] = s; }
        __syncthreads(); l2norm(u1, 96, red);
        for (int i = t; i < 96; i += 256) { int o = i/3, q = i%3; float s = 0.f;
            for (int c = 0; c < 32; ++c) for (int p = 0; p < 3; ++p)
                s += k[((o*32+c)*3+p)*3+q] * u2[c*3+p];
            v2[i] = s; }
        __syncthreads(); l2norm(v2, 96, red);
        for (int i = t; i < 96; i += 256) { int c = i/3, p = i%3; float s = 0.f;
            for (int o = 0; o < 32; ++o) for (int q = 0; q < 3; ++q)
                s += k[((o*32+c)*3+p)*3+q] * v2[o*3+q];
            u2[i] = s; }
        __syncthreads(); l2norm(u2, 96, red);
        for (int i = t; i < 32; i += 256) { float s = 0.f;
            for (int j = 0; j < 288; ++j) s += k[i*288 + j] * u3[j];
            v3[i] = s; }
        __syncthreads(); l2norm(v3, 32, red);
        for (int i = t; i < 288; i += 256) { float s = 0.f;
            for (int o = 0; o < 32; ++o) s += k[o*288 + i] * v3[o];
            u3[i] = s; }
        __syncthreads(); l2norm(u3, 288, red);
        for (int i = t; i < 32; i += 256) { float s = 0.f;
            for (int o = 0; o < 32; ++o) for (int j = 0; j < 9; ++j)
                s += k[(o*32+i)*9 + j] * u4[o*9 + j];
            v4[i] = s; }
        __syncthreads(); l2norm(v4, 32, red);
        for (int i = t; i < 288; i += 256) { int o = i/9, j = i%9; float s = 0.f;
            for (int c = 0; c < 32; ++c) s += k[(o*32+c)*9 + j] * v4[c];
            u4[i] = s; }
        __syncthreads(); l2norm(u4, 288, red);
    }
    float p1 = 0.f, p2 = 0.f, p3 = 0.f, p4 = 0.f;
    for (int i = t; i < 9216; i += 256) {
        int q = i % 3, p = (i/3) % 3, c = (i/9) % 32, o = i / 288;
        float kv = k[i];
        p1 += kv * u1[c*3+q]      * v1[o*3+p];
        p2 += kv * u2[c*3+p]      * v2[o*3+q];
        p3 += kv * u3[(c*3+p)*3+q]* v3[o];
        p4 += kv * u4[(o*3+p)*3+q]* v4[c];
    }
    float s1 = block_sum(p1, red), s2 = block_sum(p2, red);
    float s3 = block_sum(p3, red), s4 = block_sum(p4, red);
    float sigma = fminf(fminf(s1, s2), fminf(s3, s4));
    for (int i = t; i < 9216; i += 256) knout[i] = k[i] / sigma;
}

// ============== stage 3: conv-exponential (series) ================
__global__ void kg_init(const float* __restrict__ kn, float* __restrict__ kg,
                        float* __restrict__ kia) {
    int idx = blockIdx.x * 256 + threadIdx.x;
    if (idx >= CCH*CCH*KG*KG) return;
    int x = idx % 11, y = (idx/11) % 11, c = (idx/121) % 32, o = idx / (121*32);
    float v = 0.f;
    if (y >= 4 && y <= 6 && x >= 4 && x <= 6) {
        v = kn[((o*32+c)*3 + (y-4))*3 + (x-4)];
        if (o == c && y == 5 && x == 5) v += 1.f;
    }
    kg[idx] = v;
    if (idx < 9216) kia[idx] = kn[idx];
}

__global__ void convexp_step(const float* __restrict__ ki_in, const float* __restrict__ kn,
                             float* __restrict__ ki_out, float* __restrict__ kg,
                             int Sin, int iter) {
    int Sout = Sin + 2;
    int idx = blockIdx.x * 256 + threadIdx.x;
    if (idx >= 32*32*Sout*Sout) return;
    int x = idx % Sout, y = (idx/Sout) % Sout;
    int d = (idx/(Sout*Sout)) % 32, o = idx/(Sout*Sout*32);
    float s = 0.f;
    for (int c = 0; c < 32; ++c)
        for (int p = 0; p < 3; ++p) { int yi = y - p; if (yi < 0 || yi >= Sin) continue;
            for (int q = 0; q < 3; ++q) { int xi = x - q; if (xi < 0 || xi >= Sin) continue;
                s += ki_in[((c*32+d)*Sin + yi)*Sin + xi] * kn[((o*32+c)*3 + p)*3 + q]; } }
    s /= (float)iter;
    ki_out[((o*32+d)*Sout + y)*Sout + x] = s;
    int off = 5 - iter;  // center ki (size 2*iter+1) in the 11x11 kg
    kg[((o*32+d)*11 + (y+off))*11 + (x+off)] += s;
}

// ====== stage 4: pack weights into WMMA A-fragment layout (f16) ====
// A f16 16x32 layout: lane<16 -> K 0-7 (v0-3), 16-23 (v4-7); lane>=16 -> K 8-15, 24-31
__global__ void pack_w(const float* __restrict__ kg, _Float16* __restrict__ pw) {
    int idx = blockIdx.x * 256 + threadIdx.x;
    if (idx >= 121*2*32*16) return;
    int j = idx & 15, lane = (idx >> 4) & 31, half = (idx >> 9) & 1, t = idx >> 10;
    int v = j >> 1, h = j & 1;
    int hs = lane >> 4, m = lane & 15;
    int k = (v < 4) ? (8*hs + 2*v + h) : (16 + 8*hs + 2*(v-4) + h);
    int co = half*16 + m;
    int p = t / 11, q = t % 11;
    pw[idx] = (_Float16)kg[((co*32 + k)*11 + p)*11 + q];
}

// ========= stage 5: NCHW fp32 -> NHWC f16 transpose of x ==========
__global__ __launch_bounds__(256) void xpose(const float* __restrict__ x,
                                             _Float16* __restrict__ xt) {
    __shared__ float tile[32][65];
    int blk = blockIdx.x;                 // 16*256*4
    int x4 = blk & 3, y = (blk >> 2) & 255, b = blk >> 10;
    int x0 = x4 * 64, t = threadIdx.x;
    for (int r = 0; r < 8; ++r) {
        int ci = r*4 + (t >> 6), px = t & 63;
        tile[ci][px] = x[(((b*32 + ci)*256 + y) << 8) + x0 + px];
    }
    __syncthreads();
    for (int r = 0; r < 8; ++r) {
        int px = r*8 + (t >> 5), ci = t & 31;
        xt[((size_t)((b*256 + y)*256 + x0 + px))*32 + ci] = (_Float16)tile[ci][px];
    }
}

// ============ stage 6: main conv via v_wmma_f32_16x16x32_f16 =======
// Block = 8 waves = 128 contiguous output pixels of one row, all 32 out-channels.
// Per p-row: async-stage the 138-pixel x 32-ci f16 strip into LDS (80B pixel
// stride -> conflict-free 16B/lane ds_load_b128), then each wave runs 11 q-taps
// from LDS. A-fragments (wave-uniform addresses) come from global/WGP$.
#define STRIP_PX   138
#define PX_STRIDE  40          // halves per pixel slot (80 bytes)
__global__ __launch_bounds__(256) void conv_main(const _Float16* __restrict__ xt,
        const _Float16* __restrict__ pw, const float* __restrict__ bias,
        float* __restrict__ out) {
    __shared__ _Float16 sb[144 * PX_STRIDE];   // 11.25 KB
    int tid  = threadIdx.x;
    int blk  = blockIdx.x;                     // 16*256*2 = 8192
    int xh = blk & 1, y = (blk >> 1) & 255, b = blk >> 9;
    int x0 = xh * 128;
    int w = tid >> 5, lane = tid & 31;
    int n = lane & 15, hs = lane >> 4;
    v8f acc0 = {}; v8f acc1 = {};

#pragma unroll 1
    for (int p = 0; p < 11; ++p) {
        int yy = (y + p - 5) & 255;
        const _Float16* xrow = xt + ((size_t)((b*256 + yy) << 8)) * 32;

        // ---- async stage: 138 pixels * 4 x 16B chunks = 552 transfers ----
        for (int i = tid; i < STRIP_PX * 4; i += 256) {
            int pix = i >> 2, ch = i & 3;
            int px = (x0 + pix - 5) & 255;
            const _Float16* g = xrow + px*32 + ch*8;
            unsigned int loff = (unsigned int)(unsigned long long)(&sb[pix*PX_STRIDE + ch*8]);
            asm volatile("global_load_async_to_lds_b128 %0, %1, off"
                         :: "v"(loff), "v"((unsigned long long)g) : "memory");
        }
#if __has_builtin(__builtin_amdgcn_s_wait_asynccnt)
        __builtin_amdgcn_s_wait_asynccnt(0);
#else
        asm volatile("s_wait_asynccnt 0x0" ::: "memory");
#endif
        __syncthreads();

        // ---- compute 11 q-taps from LDS ----
        const _Float16* sbase = &sb[(w*16 + n) * PX_STRIDE + hs*8];
        int t0 = p * 11;
#pragma unroll
        for (int q = 0; q < 11; ++q) {
            const _Float16* bp = sbase + q * PX_STRIDE;
            v8h blo = *(const v8h*)(bp);
            v8h bhi = *(const v8h*)(bp + 16);
            v16h bf = __builtin_shufflevector(blo, bhi, 0,1,2,3,4,5,6,7,8,9,10,11,12,13,14,15);

            int t = t0 + q;
            const _Float16* ap0 = pw + (((size_t)(t*2 + 0)*32 + lane) << 4);
            const _Float16* ap1 = pw + (((size_t)(t*2 + 1)*32 + lane) << 4);
            v8h a0l = *(const v8h*)(ap0);
            v8h a0h = *(const v8h*)(ap0 + 8);
            v8h a1l = *(const v8h*)(ap1);
            v8h a1h = *(const v8h*)(ap1 + 8);
            v16h a0 = __builtin_shufflevector(a0l, a0h, 0,1,2,3,4,5,6,7,8,9,10,11,12,13,14,15);
            v16h a1 = __builtin_shufflevector(a1l, a1h, 0,1,2,3,4,5,6,7,8,9,10,11,12,13,14,15);

            acc0 = __builtin_amdgcn_wmma_f32_16x16x32_f16(false, a0, false, bf,
                                                          (short)0, acc0, false, false);
            acc1 = __builtin_amdgcn_wmma_f32_16x16x32_f16(false, a1, false, bf,
                                                          (short)0, acc1, false, false);
        }
        __syncthreads();   // protect LDS before next p-row overwrites
    }

    // store: D layout -> lane l, vgpr v : row m = v + 8*(l>=16), col n = l%16
    size_t obase = ((size_t)b*32)*65536 + (size_t)y*256 + x0 + w*16 + n;
#pragma unroll
    for (int v = 0; v < 8; ++v) {
        int co0 = v + 8*hs;
        int co1 = 16 + co0;
        out[obase + (size_t)co0*65536] = acc0[v] + bias[co0];
        out[obase + (size_t)co1*65536] = acc1[v] + bias[co1];
    }
}

// ============================ launch ==============================
extern "C" void kernel_launch(void* const* d_in, const int* in_sizes, int n_in,
                              void* d_out, int out_size, void* d_ws, size_t ws_size,
                              hipStream_t stream) {
    const float* x    = (const float*)d_in[0];
    const float* wraw = (const float*)d_in[1];
    const float* bias = (const float*)d_in[2];
    const float* u1   = (const float*)d_in[3];
    const float* u2   = (const float*)d_in[4];
    const float* u3   = (const float*)d_in[5];
    const float* u4   = (const float*)d_in[6];
    float* out = (float*)d_out;

    float* ws  = (float*)d_ws;
    float* kn  = ws + OFF_KN;
    float* kia = ws + OFF_KIA;
    float* kib = ws + OFF_KIB;
    float* kg  = ws + OFF_KG;
    _Float16* pw = (_Float16*)((char*)d_ws + PW_BYTE_OFF);
    _Float16* xt = (_Float16*)((char*)d_ws + XT_BYTE_OFF);

    skew_k      <<<36,    256, 0, stream>>>(wraw, kia);
    ff_kernel   <<<1,     256, 0, stream>>>(kia, u1, u2, u3, u4, kn);
    kg_init     <<<484,   256, 0, stream>>>(kn, kg, kia);
    convexp_step<<<100,   256, 0, stream>>>(kia, kn, kib, kg, 3, 2);
    convexp_step<<<196,   256, 0, stream>>>(kib, kn, kia, kg, 5, 3);
    convexp_step<<<324,   256, 0, stream>>>(kia, kn, kib, kg, 7, 4);
    convexp_step<<<484,   256, 0, stream>>>(kib, kn, kia, kg, 9, 5);
    pack_w      <<<484,   256, 0, stream>>>(kg, pw);
    xpose       <<<16384, 256, 0, stream>>>(x, xt);
    conv_main   <<<8192,  256, 0, stream>>>(xt, pw, bias, out);
}
